// LSTM_43233140801648
// MI455X (gfx1250) — compile-verified, matrix operand order checked by reading, not run on hardware
//
#include <hip/hip_runtime.h>
#include <hip/hip_bf16.h>
#include <math.h>

typedef __attribute__((ext_vector_type(16))) _Float16 v16h;
typedef __attribute__((ext_vector_type(8)))  _Float16 v8h;
typedef __attribute__((ext_vector_type(8)))  float    v8f;

#define Hdim 256
#define Idim 32
#define Odim 10
#define Sdim 512
#define KTOT 288      // H + I (augmented K for fused x-projection)
#define BT   16       // batch rows per workgroup (one WMMA M-tile)
#define ASTRIDE 296   // padded LDS row stride in halves (avoid bank conflicts)
#define ZSTRIDE 1028  // padded z row stride in floats

// ---------------------------------------------------------------------------
// Prep: build WcombT[n][k] = f16 of concat([Wgh;Wih;Wfh;Woh], [Wgx;Wix;Wfx;Wox])
// transposed so that for a fixed output column n, K is contiguous (two b128
// loads per lane per 32x16 B-fragment).
// ---------------------------------------------------------------------------
__global__ void prep_weights(const float* __restrict__ Wgx, const float* __restrict__ Wgh,
                             const float* __restrict__ Wix, const float* __restrict__ Wih,
                             const float* __restrict__ Wfx, const float* __restrict__ Wfh,
                             const float* __restrict__ Wox, const float* __restrict__ Woh,
                             _Float16* __restrict__ WcombT) {
    int idx = blockIdx.x * blockDim.x + threadIdx.x;   // 0 .. 1024*288-1
    if (idx >= 1024 * KTOT) return;
    int n = idx / KTOT;
    int k = idx % KTOT;
    int gate = n >> 8;
    int c    = n & 255;
    const float* Wh[4] = {Wgh, Wih, Wfh, Woh};
    const float* Wx[4] = {Wgx, Wix, Wfx, Wox};
    float v = (k < Hdim) ? Wh[gate][k * Hdim + c]
                         : Wx[gate][(k - Hdim) * Hdim + c];
    WcombT[(size_t)n * KTOT + k] = (_Float16)v;
}

// ---------------------------------------------------------------------------
// Main persistent LSTM kernel: 16 blocks x 512 threads (16 wave32, 4/SIMD).
// Block b owns batch rows [16b, 16b+16) and runs all 512 timesteps.
//
// Hybrid weight placement under the 256-VGPR/wave cap:
//   - N-tiles {4w, 4w+1}: B fragments register-resident (144 VGPRs).
//   - N-tiles {4w+2, 4w+3}: streamed from L2-resident WcombT each step via
//     global_load_b128 (a laundered 32-bit zero offset defeats LICM without
//     destroying the global address space, so no flat_load / no spill).
// Per step/wave: 18 ds_load_b128 (A panel), 36 global_load_b128 (streamed B),
// 36 v_wmma_f32_16x16x32_f16.
// ---------------------------------------------------------------------------
__launch_bounds__(512, 1)
__global__ void lstm_main(const float* __restrict__ x,
                          const _Float16* __restrict__ WcombT,
                          const float* __restrict__ bg, const float* __restrict__ bi,
                          const float* __restrict__ bf, const float* __restrict__ bo,
                          const float* __restrict__ Wph, const float* __restrict__ bp,
                          float* __restrict__ out) {
    __shared__ __align__(16) _Float16 Alds[BT * ASTRIDE];   // [h | x_t] panel, f16
    __shared__ __align__(16) float    Zlds[BT * ZSTRIDE];   // gate pre-activations
    __shared__ __align__(16) float    Clds[BT * Hdim];      // cell state

    const int tid  = threadIdx.x;
    const int wave = tid >> 5;        // 0..15
    const int lane = tid & 31;
    const int mrow = lane & 15;       // M (A) / N (B) index within tile
    const int sel  = lane >> 4;       // K-half selector per ISA fragment layout
    const int b0   = blockIdx.x * BT;

    // Zero h-region of A panel and cell state (h0 = c0 = 0).
    for (int p = 0; p < 8; ++p) {
        int idx = tid + p * 512;                 // 0..4095
        int m = idx >> 8, j = idx & 255;
        Alds[m * ASTRIDE + j] = (_Float16)0.0f;
        Clds[idx] = 0.0f;
    }

    // Per-tile output columns and biases (constant over time).
    int   nb[4];
    float bias[4];
    #pragma unroll
    for (int j = 0; j < 4; ++j) {
        nb[j] = (wave * 4 + j) * 16 + mrow;
        const float* bsel = (nb[j] < 256) ? bg : (nb[j] < 512) ? bi
                          : (nb[j] < 768) ? bf : bo;
        bias[j] = bsel[nb[j] & 255];
    }

    // ---- Register-resident B fragments for tiles j = 0,1 (144 VGPRs) ----
    // Fragment layout per ISA: lane holds column nb, K {kb..kb+7, kb+16..kb+23}
    // with kb = 32*kt + 8*sel.
    v16h Bres[2][9];
    #pragma unroll
    for (int j = 0; j < 2; ++j) {
        const _Float16* bptr = WcombT + (size_t)nb[j] * KTOT;
        #pragma unroll
        for (int kt = 0; kt < 9; ++kt) {
            const int kb = kt * 32 + sel * 8;
            v8h q0 = *(const v8h*)&bptr[kb];
            v8h q1 = *(const v8h*)&bptr[kb + 16];
            Bres[j][kt] = __builtin_shufflevector(
                q0, q1, 0,1,2,3,4,5,6,7,8,9,10,11,12,13,14,15);
        }
    }

    // 32-bit element offsets (within WcombT) for the streamed tiles.
    const unsigned int sbase2 = (unsigned int)nb[2] * KTOT;
    const unsigned int sbase3 = (unsigned int)nb[3] * KTOT;
    unsigned int zoff = 0;   // always 0 at runtime; laundered per-iteration

    for (int t = 0; t < Sdim; ++t) {
        // Opaque per-iteration offset: streamed B loads cannot be hoisted,
        // but the base pointer keeps its global address space.
        asm volatile("" : "+v"(zoff));

        // Stage x_t (16 rows x 32 cols) into A panel, f16. Exactly 512 threads.
        {
            int m = tid >> 5, i = tid & 31;
            Alds[m * ASTRIDE + Hdim + i] =
                (_Float16)x[((size_t)(b0 + m) * Sdim + t) * Idim + i];
        }
        __syncthreads();

        // z-tile GEMM: each A fragment loaded once, feeds all 4 N-tiles.
        v8f acc[4];
        #pragma unroll
        for (int j = 0; j < 4; ++j)
            acc[j] = (v8f){bias[j], bias[j], bias[j], bias[j],
                           bias[j], bias[j], bias[j], bias[j]};
        #pragma unroll
        for (int kt = 0; kt < 9; ++kt) {
            const int kb = kt * 32 + sel * 8;
            v8h a0 = *(const v8h*)&Alds[mrow * ASTRIDE + kb];
            v8h a1 = *(const v8h*)&Alds[mrow * ASTRIDE + kb + 16];
            v16h av = __builtin_shufflevector(
                a0, a1, 0,1,2,3,4,5,6,7,8,9,10,11,12,13,14,15);
            // Resident tiles
            acc[0] = __builtin_amdgcn_wmma_f32_16x16x32_f16(
                false, av, false, Bres[0][kt], (short)0, acc[0], false, false);
            acc[1] = __builtin_amdgcn_wmma_f32_16x16x32_f16(
                false, av, false, Bres[1][kt], (short)0, acc[1], false, false);
            // Streamed tiles (fresh global_load_b128 from L2 each step)
            {
                const _Float16* sp = WcombT + (sbase2 + (unsigned)kb + zoff);
                v8h q0 = *(const v8h*)sp;
                v8h q1 = *(const v8h*)(sp + 16);
                v16h bv = __builtin_shufflevector(
                    q0, q1, 0,1,2,3,4,5,6,7,8,9,10,11,12,13,14,15);
                acc[2] = __builtin_amdgcn_wmma_f32_16x16x32_f16(
                    false, av, false, bv, (short)0, acc[2], false, false);
            }
            {
                const _Float16* sp = WcombT + (sbase3 + (unsigned)kb + zoff);
                v8h q0 = *(const v8h*)sp;
                v8h q1 = *(const v8h*)(sp + 16);
                v16h bv = __builtin_shufflevector(
                    q0, q1, 0,1,2,3,4,5,6,7,8,9,10,11,12,13,14,15);
                acc[3] = __builtin_amdgcn_wmma_f32_16x16x32_f16(
                    false, av, false, bv, (short)0, acc[3], false, false);
            }
        }
        // Scatter D tiles to Zlds: VGPR r holds row (r + 8*sel), col = nb.
        #pragma unroll
        for (int j = 0; j < 4; ++j) {
            #pragma unroll
            for (int r = 0; r < 8; ++r)
                Zlds[(r + sel * 8) * ZSTRIDE + nb[j]] = acc[j][r];
        }
        __syncthreads();

        // Gate nonlinearities: 16x256 cells, 8 per thread. Same thread owns the
        // same cells every step -> no cross-thread dependence on Clds.
        #pragma unroll
        for (int p = 0; p < 8; ++p) {
            int idx = tid + p * 512;
            int m = idx >> 8, jj = idx & 255;
            const float* zr = &Zlds[m * ZSTRIDE];
            float zg = zr[jj];
            float zi = zr[256 + jj];
            float zf = zr[512 + jj];
            float zo = zr[768 + jj];
            float g  = tanhf(zg);
            float ig = 1.0f / (1.0f + __expf(-zi));
            float fg = 1.0f / (1.0f + __expf(-zf));
            float og = 1.0f / (1.0f + __expf(-zo));
            float cn = g * ig + Clds[idx] * fg;
            Clds[idx] = cn;
            Alds[m * ASTRIDE + jj] = (_Float16)(tanhf(cn) * og);
        }
        // No barrier needed here: stage-x writes a disjoint A region, and the
        // __syncthreads() after staging orders gates(t) vs wmma(t+1).
    }
    __syncthreads();

    // Final projection: logits = h @ Wph + bp  (16 rows x 10 outputs)
    if (tid < BT * Odim) {
        int m = tid / Odim, o = tid % Odim;
        float s = bp[o];
        for (int k = 0; k < Hdim; ++k)
            s += (float)Alds[m * ASTRIDE + k] * Wph[k * Odim + o];
        Zlds[m * 16 + o] = s;
    }
    __syncthreads();

    // Softmax per row, write output.
    if (tid < BT) {
        int m = tid;
        float mx = -1e30f;
        for (int o = 0; o < Odim; ++o) mx = fmaxf(mx, Zlds[m * 16 + o]);
        float e[Odim], sum = 0.0f;
        for (int o = 0; o < Odim; ++o) { e[o] = __expf(Zlds[m * 16 + o] - mx); sum += e[o]; }
        float inv = 1.0f / sum;
        for (int o = 0; o < Odim; ++o) out[(size_t)(b0 + m) * Odim + o] = e[o] * inv;
    }
}

extern "C" void kernel_launch(void* const* d_in, const int* in_sizes, int n_in,
                              void* d_out, int out_size, void* d_ws, size_t ws_size,
                              hipStream_t stream) {
    const float* x   = (const float*)d_in[0];
    const float* Wgx = (const float*)d_in[1];
    const float* Wgh = (const float*)d_in[2];
    const float* bg  = (const float*)d_in[3];
    const float* Wix = (const float*)d_in[4];
    const float* Wih = (const float*)d_in[5];
    const float* bi  = (const float*)d_in[6];
    const float* Wfx = (const float*)d_in[7];
    const float* Wfh = (const float*)d_in[8];
    const float* bf  = (const float*)d_in[9];
    const float* Wox = (const float*)d_in[10];
    const float* Woh = (const float*)d_in[11];
    const float* bo  = (const float*)d_in[12];
    const float* Wph = (const float*)d_in[13];
    const float* bp  = (const float*)d_in[14];

    _Float16* WcombT = (_Float16*)d_ws;   // 1024*288*2 = 576 KB of scratch

    int prep_elems = 1024 * KTOT;
    prep_weights<<<(prep_elems + 255) / 256, 256, 0, stream>>>(
        Wgx, Wgh, Wix, Wih, Wfx, Wfh, Wox, Woh, WcombT);

    lstm_main<<<256 / BT, 512, 0, stream>>>(
        x, WcombT, bg, bi, bf, bo, Wph, bp, (float*)d_out);
}
